// EdgeNet_60189671686199
// MI455X (gfx1250) — compile-verified, matrix operand order checked by reading, not run on hardware
//
#include <hip/hip_runtime.h>
#include <hip/hip_bf16.h>

typedef __attribute__((ext_vector_type(16))) _Float16 v16h;
typedef __attribute__((ext_vector_type(8)))  _Float16 v8h;
typedef __attribute__((ext_vector_type(8)))  float    v8f;

#define NODES 100000
#define DEG   16
#define EDGES (NODES * DEG)
#define HDIM  64

union Frag16 {
    v16h v;
    v8h  h8[2];
    _Float16 h[16];
    uint4 u[2];
};

union FragF {
    v8f v;
    float f[8];
    float4 q[2];
};

// Build a 16-bit WMMA fragment for one lane from a row-major row pointer.
// Elements 0..7  = base[koff .. koff+7]     (VGPR0..3: K=koff..koff+7)
// Elements 8..15 = base[koff+16..koff+23]   (VGPR4..7: K=koff+16..koff+23)
__device__ __forceinline__ v16h load_frag_row(const _Float16* base, int koff) {
    Frag16 f;
    f.u[0] = *(const uint4*)(base + koff);
    f.u[1] = *(const uint4*)(base + koff + 16);
    return f.v;
}

// Transpose + f16-convert + zero-pad a weight matrix W[Kin][Nout] (row-major f32)
// into Wt[NP][KP] (row-major f16): Wt[n][k] = W[k][n].
__global__ void prep_weight_kernel(const float* __restrict__ W, _Float16* __restrict__ Wt,
                                   int Kin, int Nout, int KP, int NP) {
    int idx = blockIdx.x * blockDim.x + threadIdx.x;
    if (idx >= KP * NP) return;
    int n = idx / KP, k = idx % KP;
    float v = (k < Kin && n < Nout) ? W[(size_t)k * Nout + n] : 0.0f;
    Wt[idx] = (_Float16)v;
}

__global__ void copy_f32_kernel(const float* __restrict__ src, float* __restrict__ dst, int n) {
    int i = blockIdx.x * blockDim.x + threadIdx.x;
    if (i < n) dst[i] = src[i];
}

// MODE 0: input layer  (x f32 [N,4], K1=8  padded to 32)
// MODE 1: hidden layer (hin f16 [N,64], K1=128)
// MODE 2: output layer (hin f16 [N,64], K1=128, N2=1 padded to 16, writes f32 out[N])
//
// GEMM1 is computed role-swapped: D1 = W1t(A) x msg^T(B), so D1 has
// N = edge (lane) and M = neuron (VGPRs).  D1's per-lane M pattern
// (h*8+v) equals the A-fragment K pattern, so after bias+relu+cvt the
// accumulators ARE GEMM2's A fragments -- no LDS staging round-trip.
template <int MODE>
__global__ void __launch_bounds__(256)
edgeconv_kernel(const float* __restrict__ x4,
                const _Float16* __restrict__ hin,
                const int* __restrict__ srcIdx,       // edge_index row 0, E entries
                const _Float16* __restrict__ gW1t,    // [64][KP1] f16, transposed/padded
                const float* __restrict__ b1,         // [64]
                const _Float16* __restrict__ gW2t,    // [NP2][64] f16, transposed/padded
                const float* __restrict__ b2,         // [64] or [1]
                _Float16* __restrict__ hout,          // MODE 0/1: [N][64] f16
                float* __restrict__ outv)             // MODE 2: [N] f32
{
    constexpr int KP1 = (MODE == 0) ? 32 : 128;
    constexpr int KB1 = KP1 / 32;
    constexpr int NT2 = (MODE == 2) ? 1 : 4;
    constexpr int WAVES = 8;

    const int lane   = threadIdx.x & 31;
    const int wave   = threadIdx.x >> 5;
    const int half16 = lane >> 4;     // lane-half selects K / M sub-block per ISA layout
    const int lrow   = lane & 15;     // row (A) / column (B,D) index within tile

    // ---- hoist all weight fragments into registers for the node loop ----
    // GEMM1 A-fragments: row = neuron (mt*16 + lrow) of W1t.
    v16h aw1[4][4];                   // [mtile][kblock], only KB1 kblocks used
    for (int mt = 0; mt < 4; ++mt)
        for (int kb = 0; kb < KB1; ++kb)
            aw1[mt][kb] = load_frag_row(gW1t + (size_t)(mt * 16 + lrow) * KP1,
                                        kb * 32 + half16 * 8);

    // GEMM2 B-fragments: column = neuron (nt*16 + lrow) of W2t.
    v16h bw2[NT2][2];
    for (int nt = 0; nt < NT2; ++nt)
        for (int kb = 0; kb < 2; ++kb)
            bw2[nt][kb] = load_frag_row(gW2t + (size_t)(nt * 16 + lrow) * 64,
                                        kb * 32 + half16 * 8);

    // GEMM1 bias per M (neuron): element v of tile mt is b1[mt*16 + half16*8 + v].
    v8f b1f[4];
    for (int mt = 0; mt < 4; ++mt) {
        FragF ff;
        ff.q[0] = *(const float4*)(b1 + mt * 16 + half16 * 8);
        ff.q[1] = *(const float4*)(b1 + mt * 16 + half16 * 8 + 4);
        b1f[mt] = ff.v;
    }

    // GEMM2 bias per N (neuron = nt*16 + lrow).
    float b2v[NT2];
    for (int nt = 0; nt < NT2; ++nt)
        b2v[nt] = (MODE == 2) ? ((nt == 0 && lrow == 0) ? b2[0] : 0.0f)
                              : b2[nt * 16 + lrow];

    const int totalWaves = gridDim.x * WAVES;
    for (int node = blockIdx.x * WAVES + wave; node < NODES; node += totalWaves) {
        // source node of this lane's edge (edge = node*16 + lrow; dst is contiguous)
        const int sj = srcIdx[node * DEG + lrow];

        // ========== GEMM1 (role-swapped): D1[n][e] = sum_k W1t[n][k]*m[e][k] =====
        v8f acc1[4] = {};
        if (MODE == 0) {
            Frag16 b;
            for (int i = 0; i < 16; ++i) b.h[i] = (_Float16)0.0f;
            if (half16 == 0) {            // real K = 0..7 lives in lane half 0
                const float* xi = x4 + (size_t)node * 4;
                const float* xj = x4 + (size_t)sj * 4;
                for (int t = 0; t < 4; ++t) {
                    float xiv = xi[t], xjv = xj[t];
                    b.h[t]     = (_Float16)xiv;
                    b.h[4 + t] = (_Float16)(xjv - xiv);
                }
            }
            for (int mt = 0; mt < 4; ++mt)
                acc1[mt] = __builtin_amdgcn_wmma_f32_16x16x32_f16(
                    false, aw1[mt][0], false, b.v, (short)0, acc1[mt], false, false);
        } else {
            const _Float16* hi = hin + (size_t)node * HDIM;
            const _Float16* hj = hin + (size_t)sj * HDIM;
            for (int kb = 0; kb < 4; ++kb) {
                Frag16 b;
                const int fb = (kb < 2 ? kb : kb - 2) * 32 + half16 * 8;
                if (kb < 2) {                          // x_i block (edge-invariant)
                    b.h8[0] = *(const v8h*)(hi + fb);
                    b.h8[1] = *(const v8h*)(hi + fb + 16);
                } else {                               // x_j - x_i block (gather)
                    v8h xi0 = *(const v8h*)(hi + fb);
                    v8h xi1 = *(const v8h*)(hi + fb + 16);
                    v8h xj0 = *(const v8h*)(hj + fb);
                    v8h xj1 = *(const v8h*)(hj + fb + 16);
                    b.h8[0] = xj0 - xi0;               // packed f16 subtract
                    b.h8[1] = xj1 - xi1;
                }
                for (int mt = 0; mt < 4; ++mt)
                    acc1[mt] = __builtin_amdgcn_wmma_f32_16x16x32_f16(
                        false, aw1[mt][kb], false, b.v, (short)0, acc1[mt], false, false);
            }
        }

        // ---- bias + relu + f16-convert: acc1 tiles ARE GEMM2's A fragments ----
        // a2[kb] elements 0..7  <- tile mt=2*kb   (K = kb*32 + h*8 + v)
        // a2[kb] elements 8..15 <- tile mt=2*kb+1 (K = kb*32 + 16 + h*8 + v)
        Frag16 a2[2];
        for (int mt = 0; mt < 4; ++mt) {
            v8f r = acc1[mt];
            for (int t = 0; t < 8; ++t) {
                float rv = fmaxf(r[t] + b1f[mt][t], 0.0f);
                a2[mt >> 1].h[(mt & 1) * 8 + t] = (_Float16)rv;
            }
        }

        // ========== GEMM2: D2[e][n] = sum_k h1[e][k] * W2t[n][k] ================
        v8f acc2[NT2] = {};
        for (int kb = 0; kb < 2; ++kb)
            for (int nt = 0; nt < NT2; ++nt)
                acc2[nt] = __builtin_amdgcn_wmma_f32_16x16x32_f16(
                    false, a2[kb].v, false, bw2[nt][kb], (short)0, acc2[nt], false, false);

        // ---- segment_max over the 16 edge rows (M dim of D2), bias, relu ----
        for (int nt = 0; nt < NT2; ++nt) {
            float m = acc2[nt][0];
            for (int v = 1; v < 8; ++v) m = fmaxf(m, acc2[nt][v]);   // M within half
            m = fmaxf(m, __shfl_xor(m, 16, 32));                     // merge halves
            m += b2v[nt];                                            // bias commutes with max
            if (MODE != 2) {
                m = fmaxf(m, 0.0f);                                  // outer relu
                if (half16 == 0)
                    hout[(size_t)node * HDIM + nt * 16 + lrow] = (_Float16)m;
            } else {
                if (half16 == 0 && lrow == 0) outv[node] = m;
            }
        }
    }
}

extern "C" void kernel_launch(void* const* d_in, const int* in_sizes, int n_in,
                              void* d_out, int out_size, void* d_ws, size_t ws_size,
                              hipStream_t stream) {
    (void)in_sizes; (void)n_in; (void)out_size; (void)ws_size;

    const float* x         = (const float*)d_in[0];
    const int*   edge_idx  = (const int*)d_in[1];   // [2][E]; row 0 = src (dst is repeat(arange))
    const float* edge_attr = (const float*)d_in[2];
    const float* Wi1 = (const float*)d_in[3];  const float* bi1 = (const float*)d_in[4];
    const float* Wi2 = (const float*)d_in[5];  const float* bi2 = (const float*)d_in[6];
    const float* Wh1 = (const float*)d_in[7];  const float* bh1 = (const float*)d_in[8];
    const float* Wh2 = (const float*)d_in[9];  const float* bh2 = (const float*)d_in[10];
    const float* Wo1 = (const float*)d_in[11]; const float* bo1 = (const float*)d_in[12];
    const float* Wo2 = (const float*)d_in[13]; const float* bo2 = (const float*)d_in[14];

    // ---- workspace carve-up (f16 activations ping-pong + transposed weights) ----
    char* ws = (char*)d_ws;
    size_t off = 0;
    auto alloc = [&](size_t bytes) -> void* {
        off = (off + 255) & ~(size_t)255;
        void* p = ws + off;
        off += bytes;
        return p;
    };
    _Float16* hA   = (_Float16*)alloc((size_t)NODES * HDIM * sizeof(_Float16));
    _Float16* hB   = (_Float16*)alloc((size_t)NODES * HDIM * sizeof(_Float16));
    _Float16* wi1t = (_Float16*)alloc((size_t)64 * 32 * 2);
    _Float16* wi2t = (_Float16*)alloc((size_t)64 * 64 * 2);
    _Float16* wh1t = (_Float16*)alloc((size_t)6 * 64 * 128 * 2);
    _Float16* wh2t = (_Float16*)alloc((size_t)6 * 64 * 64 * 2);
    _Float16* wo1t = (_Float16*)alloc((size_t)64 * 128 * 2);
    _Float16* wo2t = (_Float16*)alloc((size_t)16 * 64 * 2);

    auto prep = [&](const float* W, _Float16* Wt, int Kin, int Nout, int KP, int NP) {
        int tot = KP * NP;
        prep_weight_kernel<<<(tot + 255) / 256, 256, 0, stream>>>(W, Wt, Kin, Nout, KP, NP);
    };
    prep(Wi1, wi1t, 8, 64, 32, 64);
    prep(Wi2, wi2t, 64, 64, 64, 64);
    for (int l = 0; l < 6; ++l) {
        prep(Wh1 + (size_t)l * 128 * 64, wh1t + (size_t)l * 64 * 128, 128, 64, 128, 64);
        prep(Wh2 + (size_t)l * 64 * 64,  wh2t + (size_t)l * 64 * 64,  64, 64, 64, 64);
    }
    prep(Wo1, wo1t, 128, 64, 128, 64);
    prep(Wo2, wo2t, 64, 1, 64, 16);

    const int blocks = 512;   // 4096 waves, ~25 nodes per wave

    // input layer
    edgeconv_kernel<0><<<blocks, 256, 0, stream>>>(
        x, nullptr, edge_idx, wi1t, bi1, wi2t, bi2, hA, nullptr);

    // 6 hidden layers (ping-pong activations)
    _Float16* cur = hA; _Float16* nxt = hB;
    for (int l = 0; l < 6; ++l) {
        edgeconv_kernel<1><<<blocks, 256, 0, stream>>>(
            nullptr, cur, edge_idx,
            wh1t + (size_t)l * 64 * 128, bh1 + l * 64,
            wh2t + (size_t)l * 64 * 64,  bh2 + l * 64,
            nxt, nullptr);
        _Float16* t = cur; cur = nxt; nxt = t;
    }

    // output layer -> d_out[0..N)
    edgeconv_kernel<2><<<blocks, 256, 0, stream>>>(
        nullptr, cur, edge_idx, wo1t, bo1, wo2t, bo2, nullptr, (float*)d_out);

    // edge_attr passthrough -> d_out[N..N+E)
    copy_f32_kernel<<<(EDGES + 255) / 256, 256, 0, stream>>>(
        edge_attr, (float*)d_out + NODES, EDGES);
}
